// TopKSAE_44023414784639
// MI455X (gfx1250) — matrix operand order repaired and checked by reading, not compiled, not run on hardware
//
#include <hip/hip_runtime.h>
#include <hip/hip_bf16.h>

#define EMBD  768
#define FEATS 24576
#define BT    4096
#define TOPK  32

#define BM 128
#define BN 256
#define BK 32
#define KP 40           // BK + 8 pad: preserves 16B alignment of b128 ops, avoids bank conflicts

typedef __attribute__((ext_vector_type(16))) __bf16 v16bf;
typedef __attribute__((ext_vector_type(8)))  float  v8f;

union FragU { v16bf v; int4 q[2]; };                  // 32B lane fragment (A or B)

__device__ __forceinline__ unsigned short f2bf(float f) {
    unsigned u = __float_as_uint(f);
    u += 0x7FFFu + ((u >> 16) & 1u);                  // round-to-nearest-even
    return (unsigned short)(u >> 16);
}

// order-preserving float <-> uint key transform (for exact radix top-k)
__device__ __forceinline__ unsigned fkey(float f) {
    unsigned u = __float_as_uint(f);
    return (u & 0x80000000u) ? ~u : (u ^ 0x80000000u);
}
__device__ __forceinline__ float ikey(unsigned k) {
    unsigned u = (k & 0x80000000u) ? (k ^ 0x80000000u) : ~k;
    return __uint_as_float(u);
}

// -------- prep: W_enc fp32 [e][f]  ->  bf16 TRANSPOSED wbT [f][e] ------------
__global__ __launch_bounds__(256) void k_prep_wT(const float* __restrict__ w,
                                                 unsigned short* __restrict__ wbT) {
    __shared__ unsigned short tile[32][33];
    const int tx = threadIdx.x;                       // 0..31
    const int ty = threadIdx.y;                       // 0..7
    const int f0 = blockIdx.x * 32;
    const int e0 = blockIdx.y * 32;
#pragma unroll
    for (int r = ty; r < 32; r += 8)                  // read coalesced along f
        tile[r][tx] = f2bf(w[(long long)(e0 + r) * FEATS + f0 + tx]);
    __syncthreads();
#pragma unroll
    for (int r = ty; r < 32; r += 8)                  // write coalesced along e
        wbT[(long long)(f0 + r) * EMBD + e0 + tx] = tile[tx][r];
}

// -------- prep: xc = x - b_dec, fp32 -> bf16 ---------------------------------
__global__ __launch_bounds__(256) void k_prep_x(const float* __restrict__ x,
                                                const float* __restrict__ b_dec,
                                                unsigned short* __restrict__ xcb) {
    long long i = ((long long)blockIdx.x * 256 + threadIdx.x) * 4;
    int e = (int)(i % EMBD);                          // EMBD % 4 == 0, stays in-row
    float4 v = *reinterpret_cast<const float4*>(x + i);
    float4 b = *reinterpret_cast<const float4*>(b_dec + e);
    ushort4 o = { f2bf(v.x - b.x), f2bf(v.y - b.y), f2bf(v.z - b.z), f2bf(v.w - b.w) };
    *reinterpret_cast<ushort4*>(xcb + i) = o;
}

// -------- encoder GEMM: latent = xc @ W_enc + b_enc --------------------------
// C[4096][24576] fp32; A bf16 [4096][768]; Bt = W_enc^T bf16 [24576][768].
// 128x256 block tile, 8 waves each owning a 64x64 (4x4 frag) tile:
// 16 WMMA per 16 ds_load_b128 per wave per k-step. Double-buffered LDS.
__global__ __launch_bounds__(256) void k_encode(const unsigned short* __restrict__ A,
                                                const unsigned short* __restrict__ Bt,
                                                const float* __restrict__ b_enc,
                                                float* __restrict__ C) {
    __shared__ unsigned short As[2][BM * KP];         // [m][k], 10 KB each
    __shared__ unsigned short Bs[2][BN * KP];         // [n][k], 20 KB each

    const int tid   = threadIdx.x;
    const int lane  = tid & 31;
    const int wave  = tid >> 5;                       // 8 waves
    const int wm    = wave & 1;                       // 2 along M (64 rows each)
    const int wn    = wave >> 1;                      // 4 along N (64 cols each)
    const int tileM = blockIdx.y * BM;
    const int tileN = blockIdx.x * BN;

    // staging: (row, 16k-chunk) pairs; consecutive threads cover one row (64B)
    const int srow   = tid >> 1;                      // 0..127
    const int schunk = (tid & 1) * 16;                // 0 | 16
    const unsigned short* ga  = A  + (long long)(tileM + srow) * EMBD + schunk;
    const unsigned short* gb0 = Bt + (long long)(tileN + srow) * EMBD + schunk;
    const unsigned short* gb1 = gb0 + (long long)128 * EMBD;
    const int aoff  = srow * KP + schunk;
    const int boff0 = srow * KP + schunk;
    const int boff1 = (srow + 128) * KP + schunk;

    const int lhalf = lane >> 4;                      // 0 | 1
    const int l15   = lane & 15;

    v8f acc[4][4];
    const v8f vzero = {0.f,0.f,0.f,0.f,0.f,0.f,0.f,0.f};
#pragma unroll
    for (int i = 0; i < 4; ++i)
#pragma unroll
        for (int j = 0; j < 4; ++j) acc[i][j] = vzero;

    // prologue: stage tile 0
    int4 ra0  = *reinterpret_cast<const int4*>(ga);
    int4 ra1  = *reinterpret_cast<const int4*>(ga + 8);
    int4 rb00 = *reinterpret_cast<const int4*>(gb0);
    int4 rb01 = *reinterpret_cast<const int4*>(gb0 + 8);
    int4 rb10 = *reinterpret_cast<const int4*>(gb1);
    int4 rb11 = *reinterpret_cast<const int4*>(gb1 + 8);
    *reinterpret_cast<int4*>(&As[0][aoff])      = ra0;
    *reinterpret_cast<int4*>(&As[0][aoff + 8])  = ra1;
    *reinterpret_cast<int4*>(&Bs[0][boff0])     = rb00;
    *reinterpret_cast<int4*>(&Bs[0][boff0 + 8]) = rb01;
    *reinterpret_cast<int4*>(&Bs[0][boff1])     = rb10;
    *reinterpret_cast<int4*>(&Bs[0][boff1 + 8]) = rb11;
    __syncthreads();

    int buf = 0;
    for (int k0 = 0; k0 < EMBD; k0 += BK) {
        const int nxt = k0 + BK;
        if (nxt < EMBD) {                             // global fetch of next tile
            ra0  = *reinterpret_cast<const int4*>(ga  + nxt);
            ra1  = *reinterpret_cast<const int4*>(ga  + nxt + 8);
            rb00 = *reinterpret_cast<const int4*>(gb0 + nxt);
            rb01 = *reinterpret_cast<const int4*>(gb0 + nxt + 8);
            rb10 = *reinterpret_cast<const int4*>(gb1 + nxt);
            rb11 = *reinterpret_cast<const int4*>(gb1 + nxt + 8);
            if (nxt + BK < EMBD) {                    // -> global_prefetch_b8
                __builtin_prefetch(ga  + nxt + BK, 0, 3);
                __builtin_prefetch(gb0 + nxt + BK, 0, 3);
                __builtin_prefetch(gb1 + nxt + BK, 0, 3);
            }
        }

        // B fragments resident (32 VGPRs); A fragment streamed per i
        FragU bf[4];
#pragma unroll
        for (int j = 0; j < 4; ++j) {                 // B 32x16: lanes 0-15 K0..15, lanes 16-31 K16..31
            const unsigned short* p = &Bs[buf][(wn * 64 + j * 16 + l15) * KP + lhalf * 16];
            bf[j].q[0] = *reinterpret_cast<const int4*>(p);
            bf[j].q[1] = *reinterpret_cast<const int4*>(p + 8);
        }
#pragma unroll
        for (int i = 0; i < 4; ++i) {                 // A 16x32: lanes 0-15 K0..15, lanes 16-31 K16..31
            FragU af;
            const unsigned short* p = &As[buf][(wm * 64 + i * 16 + l15) * KP + lhalf * 16];
            af.q[0] = *reinterpret_cast<const int4*>(p);
            af.q[1] = *reinterpret_cast<const int4*>(p + 8);
#pragma unroll
            for (int j = 0; j < 4; ++j)
                acc[i][j] = __builtin_amdgcn_wmma_f32_16x16x32_bf16(
                    false, af.v, false, bf[j].v, (short)0, acc[i][j], false, false);
        }

        if (nxt < EMBD) {                             // stage next tile into other buffer
            *reinterpret_cast<int4*>(&As[buf ^ 1][aoff])      = ra0;
            *reinterpret_cast<int4*>(&As[buf ^ 1][aoff + 8])  = ra1;
            *reinterpret_cast<int4*>(&Bs[buf ^ 1][boff0])     = rb00;
            *reinterpret_cast<int4*>(&Bs[buf ^ 1][boff0 + 8]) = rb01;
            *reinterpret_cast<int4*>(&Bs[buf ^ 1][boff1])     = rb10;
            *reinterpret_cast<int4*>(&Bs[buf ^ 1][boff1 + 8]) = rb11;
        }
        __syncthreads();
        buf ^= 1;
    }

    // epilogue: C/D layout — VGPR r: M = r + 8*lhalf, N = l15; add b_enc
#pragma unroll
    for (int j = 0; j < 4; ++j) {
        const int col  = tileN + wn * 64 + j * 16 + l15;
        const float bias = b_enc[col];
#pragma unroll
        for (int i = 0; i < 4; ++i) {
            const int rowBase = tileM + wm * 64 + i * 16 + lhalf * 8;
#pragma unroll
            for (int r = 0; r < 8; ++r)
                C[(long long)(rowBase + r) * FEATS + col] = acc[i][j][r] + bias;
        }
    }
}

// -------- fused top-k threshold + mask + sparse decode -----------------------
#define NTHR 512
#define MAXC 256
#define NV4  (FEATS / 4)                              // 6144 uint4 per row

__global__ __launch_bounds__(NTHR) void k_topk_decode(float* __restrict__ fm,
                                                      const float* __restrict__ Wd,
                                                      const float* __restrict__ bd,
                                                      float* __restrict__ recon) {
    extern __shared__ unsigned keys[];                // FEATS uints = 96 KB of 320 KB LDS
    __shared__ unsigned s_cnt[32];
    __shared__ unsigned s_tcnt[NTHR + 1];
    __shared__ unsigned s_feat[MAXC];
    __shared__ float    s_val[MAXC];

    const int t   = blockIdx.x;
    const int tid = threadIdx.x;
    float* row = fm + (long long)t * FEATS;
    uint4*  k4 = reinterpret_cast<uint4*>(keys);
    float4* r4 = reinterpret_cast<float4*>(row);

    if (tid < 32) s_cnt[tid] = 0;
    for (int i = tid; i < NV4; i += NTHR) {           // coalesced b128 loads
        float4 v = r4[i];
        k4[i] = make_uint4(fkey(v.x), fkey(v.y), fkey(v.z), fkey(v.w));
    }
    __syncthreads();

    // exact Kth-largest via MSB-greedy search: largest T with count(key >= T) >= K
    unsigned prefix = 0;
    for (int bit = 31; bit >= 0; --bit) {
        const unsigned cand = prefix | (1u << bit);
        unsigned c = 0;
        for (int i = tid; i < NV4; i += NTHR) {
            uint4 kk = k4[i];
            c += (unsigned)(kk.x >= cand) + (unsigned)(kk.y >= cand) +
                 (unsigned)(kk.z >= cand) + (unsigned)(kk.w >= cand);
        }
        if (c) atomicAdd(&s_cnt[bit], c);
        __syncthreads();
        if (s_cnt[bit] >= TOPK) prefix = cand;        // same decision in every thread
    }
    const unsigned T = prefix;

    // mask (latent >= kth, ties included) + per-thread keep count
    unsigned c = 0;
    for (int i = tid; i < NV4; i += NTHR) {
        uint4 kk = k4[i];
        float4 o;
        o.x = (kk.x >= T) ? ikey(kk.x) : 0.f;  c += (kk.x >= T);
        o.y = (kk.y >= T) ? ikey(kk.y) : 0.f;  c += (kk.y >= T);
        o.z = (kk.z >= T) ? ikey(kk.z) : 0.f;  c += (kk.z >= T);
        o.w = (kk.w >= T) ? ikey(kk.w) : 0.f;  c += (kk.w >= T);
        r4[i] = o;                                    // coalesced b128 stores
    }
    s_tcnt[tid] = c;
    __syncthreads();
    if (tid == 0) {                                   // deterministic exclusive scan
        unsigned run = 0;
        for (int u = 0; u < NTHR; ++u) { unsigned v = s_tcnt[u]; s_tcnt[u] = run; run += v; }
        s_tcnt[NTHR] = run;
    }
    __syncthreads();

    // deterministic ordered collection of nonzeros
    unsigned off = s_tcnt[tid];
    for (int i = tid; i < NV4; i += NTHR) {
        uint4 kk = k4[i];
        const unsigned fb = (unsigned)i * 4u;
        if (kk.x >= T) { if (off < MAXC) { s_feat[off] = fb;     s_val[off] = ikey(kk.x); } ++off; }
        if (kk.y >= T) { if (off < MAXC) { s_feat[off] = fb + 1; s_val[off] = ikey(kk.y); } ++off; }
        if (kk.z >= T) { if (off < MAXC) { s_feat[off] = fb + 2; s_val[off] = ikey(kk.z); } ++off; }
        if (kk.w >= T) { if (off < MAXC) { s_feat[off] = fb + 3; s_val[off] = ikey(kk.w); } ++off; }
    }
    __syncthreads();

    // sparse decode: recon = sum fm[f] * W_dec[f][:] + b_dec   (W_dec L2-resident)
    float r0 = 0.f, r1 = 0.f;
    const int e0 = tid, e1 = tid + NTHR;
    const unsigned total = s_tcnt[NTHR];
    const unsigned m = (total < MAXC) ? total : MAXC;
    for (unsigned u = 0; u < m; ++u) {
        const float* wr = Wd + (long long)s_feat[u] * EMBD;
        const float v = s_val[u];
        r0 += v * wr[e0];
        if (e1 < EMBD) r1 += v * wr[e1];
    }
    recon[(long long)t * EMBD + e0] = r0 + bd[e0];
    if (e1 < EMBD) recon[(long long)t * EMBD + e1] = r1 + bd[e1];
}

// -------- launch -------------------------------------------------------------
extern "C" void kernel_launch(void* const* d_in, const int* in_sizes, int n_in,
                              void* d_out, int out_size, void* d_ws, size_t ws_size,
                              hipStream_t stream) {
    const float* x     = (const float*)d_in[0];
    const float* W_enc = (const float*)d_in[1];
    const float* W_dec = (const float*)d_in[2];
    const float* b_enc = (const float*)d_in[3];
    const float* b_dec = (const float*)d_in[4];

    float* recon = (float*)d_out;                       // [4096][768]
    float* fm    = recon + (size_t)BT * EMBD;           // [4096][24576]

    unsigned short* wbT = (unsigned short*)d_ws;        // bf16 W_enc^T: 37.75 MB
    unsigned short* xcb = wbT + (size_t)FEATS * EMBD;   // bf16 (x - b_dec): 6.3 MB

    k_prep_wT<<<dim3(FEATS / 32, EMBD / 32), dim3(32, 8), 0, stream>>>(W_enc, wbT);
    k_prep_x<<<(BT * EMBD) / (4 * 256), 256, 0, stream>>>(x, b_dec, xcb);

    dim3 g(FEATS / BN, BT / BM);                        // 96 x 32
    k_encode<<<g, 256, 0, stream>>>(xcb, wbT, b_enc, fm);

    k_topk_decode<<<BT, NTHR, FEATS * sizeof(unsigned), stream>>>(fm, W_dec, b_dec, recon);
}
